// Qwen35VLVisionAttention_12146167513351
// MI455X (gfx1250) — matrix operand order repaired and checked by reading, not compiled, not run on hardware
//
#include <hip/hip_runtime.h>
#include <math.h>

namespace {

constexpr int kN    = 2048;
constexpr int kDim  = 1280;
constexpr int kH    = 16;
constexpr int kHd   = 80;
constexpr int kDp   = 96;    // head dim padded to 3 x 32 for bf16 WMMA K-chunks
constexpr int kNqkv = 3840;

constexpr int kKStride = 104; // LDS K-tile row stride (elems): 208B, 16B-aligned, conflict-free
constexpr int kVStride = 40;  // LDS V-tile row stride (elems): 80B,  16B-aligned, conflict-free

typedef __bf16 bf16;
typedef __attribute__((ext_vector_type(16))) __bf16 v16bf;
typedef __attribute__((ext_vector_type(8)))  __bf16 v8bf;
typedef __attribute__((ext_vector_type(8)))  float  v8f;

__device__ __forceinline__ v8f wmma_bf16(v16bf a, v16bf b, v8f c) {
  // D = A(16x32 bf16) x B(32x16 bf16) + C(16x16 f32)
  return __builtin_amdgcn_wmma_f32_16x16x32_bf16(false, a, false, b, (short)0, c,
                                                 false, false);
}

// A-fragment (16x32): lane holds row (lane%16) of a row-major tile.
// lanes 0-15: K = [0..7] ++ [16..23]; lanes 16-31: K = [8..15] ++ [24..31].
__device__ __forceinline__ v16bf load_frag_a(const bf16* p, int ld) {
  const int lane = threadIdx.x & 31;
  const bf16* q = p + (lane & 15) * ld + ((lane & 16) ? 8 : 0);
  v8bf lo = *(const v8bf*)(q);
  v8bf hi = *(const v8bf*)(q + 16);
  return __builtin_shufflevector(lo, hi, 0, 1, 2, 3, 4, 5, 6, 7,
                                 8, 9, 10, 11, 12, 13, 14, 15);
}

// B-fragment (32x16) read from K-major ("Bt", N x K row-major) storage:
// lane holds column N=(lane%16) of B == row (lane%16) of Bt.
// lanes 0-15: K = [0..7] ++ [16..23]; lanes 16-31: K = [8..15] ++ [24..31].
__device__ __forceinline__ v16bf load_frag_b(const bf16* p, int ld) {
  const int lane = threadIdx.x & 31;
  const bf16* q = p + (lane & 15) * ld + ((lane & 16) ? 8 : 0);
  v8bf lo = *(const v8bf*)(q);
  v8bf hi = *(const v8bf*)(q + 16);
  return __builtin_shufflevector(lo, hi, 0, 1, 2, 3, 4, 5, 6, 7,
                                 8, 9, 10, 11, 12, 13, 14, 15);
}

// ---------------- fp32 -> bf16 conversion helpers ----------------

__global__ void cvt_bf16_kernel(const float* __restrict__ x, bf16* __restrict__ y, int n) {
  const int i = blockIdx.x * blockDim.x + threadIdx.x;
  if (i < n) y[i] = (bf16)x[i];
}

// w: K x Ncols (row-major fp32) -> wt: Ncols x K (row-major bf16)
__global__ void transpose_bf16_kernel(const float* __restrict__ w, bf16* __restrict__ wt,
                                      int K, int Ncols) {
  const int i = blockIdx.x * blockDim.x + threadIdx.x;
  if (i < K * Ncols) {
    const int k = i / Ncols;
    const int n = i % Ncols;
    wt[(size_t)n * K + k] = (bf16)w[i];
  }
}

// -------- bf16 WMMA GEMM: C = A @ Bt^T + bias (f32 out), 32x64 per wave --------
// A: M x K bf16 row-major. Bt: N x K bf16 row-major. Software-pipelined K loop:
// fragments for chunk k+32 are loaded before the 8 WMMAs of chunk k issue.
__global__ void gemm_bf16_kernel(const bf16* __restrict__ A, const bf16* __restrict__ Bt,
                                 const float* __restrict__ bias, float* __restrict__ C,
                                 int M, int Ncols, int K) {
  const int wave = blockIdx.x * (blockDim.x >> 5) + (threadIdx.x >> 5);
  const int lane = threadIdx.x & 31;
  const int ntiles = Ncols >> 6;
  const int mt = wave / ntiles;
  const int nt = wave % ntiles;
  if (mt * 32 >= M) return;  // wave-uniform: EXEC stays all-ones for WMMA

  const bf16* Ar0 = A + (size_t)mt * 32 * K;
  const bf16* Ar1 = Ar0 + (size_t)16 * K;
  const bf16* Bp  = Bt + (size_t)nt * 64 * K;

  v8f acc[2][4] = {};
  v16bf a0 = load_frag_a(Ar0, K);
  v16bf a1 = load_frag_a(Ar1, K);
  v16bf b0 = load_frag_b(Bp + (size_t)0 * 16 * K, K);
  v16bf b1 = load_frag_b(Bp + (size_t)1 * 16 * K, K);
  v16bf b2 = load_frag_b(Bp + (size_t)2 * 16 * K, K);
  v16bf b3 = load_frag_b(Bp + (size_t)3 * 16 * K, K);

  for (int k = 32; k < K; k += 32) {
    const v16bf na0 = load_frag_a(Ar0 + k, K);
    const v16bf na1 = load_frag_a(Ar1 + k, K);
    const v16bf nb0 = load_frag_b(Bp + (size_t)0 * 16 * K + k, K);
    const v16bf nb1 = load_frag_b(Bp + (size_t)1 * 16 * K + k, K);
    const v16bf nb2 = load_frag_b(Bp + (size_t)2 * 16 * K + k, K);
    const v16bf nb3 = load_frag_b(Bp + (size_t)3 * 16 * K + k, K);
    acc[0][0] = wmma_bf16(a0, b0, acc[0][0]);
    acc[1][0] = wmma_bf16(a1, b0, acc[1][0]);
    acc[0][1] = wmma_bf16(a0, b1, acc[0][1]);
    acc[1][1] = wmma_bf16(a1, b1, acc[1][1]);
    acc[0][2] = wmma_bf16(a0, b2, acc[0][2]);
    acc[1][2] = wmma_bf16(a1, b2, acc[1][2]);
    acc[0][3] = wmma_bf16(a0, b3, acc[0][3]);
    acc[1][3] = wmma_bf16(a1, b3, acc[1][3]);
    a0 = na0; a1 = na1; b0 = nb0; b1 = nb1; b2 = nb2; b3 = nb3;
  }
  acc[0][0] = wmma_bf16(a0, b0, acc[0][0]);
  acc[1][0] = wmma_bf16(a1, b0, acc[1][0]);
  acc[0][1] = wmma_bf16(a0, b1, acc[0][1]);
  acc[1][1] = wmma_bf16(a1, b1, acc[1][1]);
  acc[0][2] = wmma_bf16(a0, b2, acc[0][2]);
  acc[1][2] = wmma_bf16(a1, b2, acc[1][2]);
  acc[0][3] = wmma_bf16(a0, b3, acc[0][3]);
  acc[1][3] = wmma_bf16(a1, b3, acc[1][3]);

#pragma unroll
  for (int s = 0; s < 2; ++s) {
    const int rb = mt * 32 + s * 16 + ((lane & 16) ? 8 : 0);
#pragma unroll
    for (int j = 0; j < 4; ++j) {
      const int n = nt * 64 + j * 16 + (lane & 15);
      const float bv = bias[n];
#pragma unroll
      for (int r = 0; r < 8; ++r)
        C[(size_t)(rb + r) * Ncols + n] = acc[s][j][r] + bv;
    }
  }
}

// ---------------- RoPE + repack into WMMA-friendly buffers ----------------
__global__ void rope_pack_kernel(const float* __restrict__ qkv,
                                 const float* __restrict__ cosb,
                                 const float* __restrict__ sinb,
                                 bf16* __restrict__ Qb, bf16* __restrict__ Kb,
                                 bf16* __restrict__ Vt) {
  const int n = blockIdx.x / kH;
  const int h = blockIdx.x % kH;
  const int d = threadIdx.x;
  const size_t qo = ((size_t)h * kN + n) * kDp + d;
  if (d < kHd) {
    const size_t base = (size_t)n * kNqkv + h * kHd;
    const float c = cosb[n * kHd + d];
    const float s = sinb[n * kHd + d];
    const float qv = qkv[base + d];
    const float kv = qkv[base + kDim + d];
    const float vv = qkv[base + 2 * kDim + d];
    const float qr = (d < kHd / 2) ? -qkv[base + d + kHd / 2]
                                   :  qkv[base + d - kHd / 2];
    const float kr = (d < kHd / 2) ? -qkv[base + kDim + d + kHd / 2]
                                   :  qkv[base + kDim + d - kHd / 2];
    Qb[qo] = (bf16)(qv * c + qr * s);
    Kb[qo] = (bf16)(kv * c + kr * s);
    Vt[((size_t)h * kHd + d) * kN + n] = (bf16)vv;
  } else {
    Qb[qo] = (bf16)0.0f;
    Kb[qo] = (bf16)0.0f;
  }
}

// ---------------- flash attention ----------------
// grid = H * (N/64) blocks of 128 threads (4 wave32 waves); each wave owns a
// 16-query tile. K/V chunks (32 keys) are shared by all 4 waves, so they are
// staged once per block into double-buffered LDS with
// GLOBAL_LOAD_ASYNC_TO_LDS_B128 (ASYNCcnt-tracked), overlapped with compute.
__global__ void attn_kernel(const bf16* __restrict__ Qb, const bf16* __restrict__ Kb,
                            const bf16* __restrict__ Vt, const int* __restrict__ cu,
                            int ncu, bf16* __restrict__ Ob) {
  __shared__ __align__(16) bf16 Kt_s[2][32][kKStride];   // keys x dims (padded)
  __shared__ __align__(16) bf16 Vt_s[2][kHd][kVStride];  // dims x keys (padded)
  __shared__ __align__(16) bf16 probs[4][16 * 32];

  const int wave = threadIdx.x >> 5;
  const int lane = threadIdx.x & 31;
  const int h  = blockIdx.x / (kN / 64);
  const int qb = blockIdx.x % (kN / 64);
  const int q0 = qb * 64 + wave * 16;

  // Segment bounds (cu_seqlens boundaries are 1024-aligned here, so all four
  // waves of a block agree -> barrier-in-loop is safe).
  int s0 = 0, s1 = kN;
  for (int i = 0; i + 1 < ncu; ++i) {
    const int a = cu[i], b = cu[i + 1];
    if (a <= q0 && q0 < b) { s0 = a; s1 = b; }
  }

  const bf16* Qh = Qb + (size_t)h * kN * kDp;
  const bf16* Kh = Kb + (size_t)h * kN * kDp;
  const bf16* Vh = Vt + (size_t)h * kHd * kN;
  bf16* pb = probs[wave];

  // Stage one 32-key chunk into LDS buffer `buf`: K tile 32x96 (384 16B units)
  // + V tile 80x32 (320 units, padded to 384 with idempotent duplicates) =
  // 768 units / 128 threads = exactly 6 async instructions per wave.
  auto stage = [&](int buf, int kk) {
    const int t = threadIdx.x;
#pragma unroll
    for (int u = 0; u < 6; ++u) {
      const int idx = t + u * 128;
      unsigned lds;
      unsigned long long g;
      if (idx < 384) {  // K tile: 32 rows x 12 16B-units
        const int row = idx / 12, cuK = idx % 12;
        lds = (unsigned)(size_t)&Kt_s[buf][row][cuK * 8];
        g   = (unsigned long long)(size_t)(Kh + (size_t)(kk + row) * kDp + cuK * 8);
      } else {          // V tile: 80 rows x 4 16B-units (+64 duplicate units)
        int vi = idx - 384;
        if (vi >= 320) vi -= 64;
        const int row = vi / 4, cuV = vi % 4;
        lds = (unsigned)(size_t)&Vt_s[buf][row][cuV * 8];
        g   = (unsigned long long)(size_t)(Vh + (size_t)row * kN + kk + cuV * 8);
      }
      asm volatile("global_load_async_to_lds_b128 %0, %1, off"
                   :: "v"(lds), "v"(g) : "memory");
    }
  };

  const v16bf qa0 = load_frag_a(Qh + (size_t)q0 * kDp + 0,  kDp);
  const v16bf qa1 = load_frag_a(Qh + (size_t)q0 * kDp + 32, kDp);
  const v16bf qa2 = load_frag_a(Qh + (size_t)q0 * kDp + 64, kDp);

  v8f acc[5] = {};
  float m[8], l[8];
#pragma unroll
  for (int r = 0; r < 8; ++r) { m[r] = -__builtin_inff(); l[r] = 0.0f; }

  const float scale = 0.11180339887498949f;  // 80^-0.5
  const int nch = (s1 - s0) >> 5;

  stage(0, s0);
  for (int i = 0; i < nch; ++i) {
    const int buf = i & 1;
    if (i + 1 < nch) {
      stage(buf ^ 1, s0 + (i + 1) * 32);
      asm volatile("s_wait_asynccnt 0x6" ::: "memory");  // chunk i's 6 ops done
    } else {
      asm volatile("s_wait_asynccnt 0x0" ::: "memory");
    }
    __syncthreads();  // all waves' staging visible

    v8f sc0 = {}, sc1 = {};
    const bf16* K0 = &Kt_s[buf][0][0];
    const bf16* K1 = &Kt_s[buf][16][0];
    sc0 = wmma_bf16(qa0, load_frag_b(K0 + 0,  kKStride), sc0);
    sc0 = wmma_bf16(qa1, load_frag_b(K0 + 32, kKStride), sc0);
    sc0 = wmma_bf16(qa2, load_frag_b(K0 + 64, kKStride), sc0);
    sc1 = wmma_bf16(qa0, load_frag_b(K1 + 0,  kKStride), sc1);
    sc1 = wmma_bf16(qa1, load_frag_b(K1 + 32, kKStride), sc1);
    sc1 = wmma_bf16(qa2, load_frag_b(K1 + 64, kKStride), sc1);

    // Online softmax: each lane owns 8 rows (C-layout), row spread over 16 lanes.
#pragma unroll
    for (int r = 0; r < 8; ++r) {
      const float a = sc0[r] * scale;
      const float b = sc1[r] * scale;
      float mx = fmaxf(a, b);
#pragma unroll
      for (int off = 8; off > 0; off >>= 1)
        mx = fmaxf(mx, __shfl_xor(mx, off, 16));
      const float mnew = fmaxf(m[r], mx);
      const float corr = __expf(m[r] - mnew);
      m[r] = mnew;
      const float p0 = __expf(a - mnew);
      const float p1 = __expf(b - mnew);
      float ps = p0 + p1;
#pragma unroll
      for (int off = 8; off > 0; off >>= 1)
        ps += __shfl_xor(ps, off, 16);
      l[r] = l[r] * corr + ps;
#pragma unroll
      for (int t = 0; t < 5; ++t) acc[t][r] *= corr;
      const int row = r + ((lane & 16) ? 8 : 0);
      pb[row * 32 + (lane & 15)]      = (bf16)p0;
      pb[row * 32 + 16 + (lane & 15)] = (bf16)p1;
    }

    asm volatile("s_wait_dscnt 0" ::: "memory");
    const v16bf pa = load_frag_a(pb, 32);
#pragma unroll
    for (int t = 0; t < 5; ++t)
      acc[t] = wmma_bf16(pa, load_frag_b(&Vt_s[buf][t * 16][0], kVStride), acc[t]);

    __syncthreads();  // all waves done reading buf before it is restaged
  }

#pragma unroll
  for (int r = 0; r < 8; ++r) {
    const float inv = 1.0f / l[r];
    const int row = q0 + r + ((lane & 16) ? 8 : 0);
    bf16* orow = Ob + (size_t)row * kDim + h * kHd + (lane & 15);
#pragma unroll
    for (int t = 0; t < 5; ++t)
      orow[t * 16] = (bf16)(acc[t][r] * inv);
  }
}

}  // namespace

extern "C" void kernel_launch(void* const* d_in, const int* in_sizes, int n_in,
                              void* d_out, int out_size, void* d_ws, size_t ws_size,
                              hipStream_t stream) {
  const float* hidden = (const float*)d_in[0];
  const int*   cu     = (const int*)d_in[1];
  const float* cosb   = (const float*)d_in[2];
  const float* sinb   = (const float*)d_in[3];
  const float* Wqkv   = (const float*)d_in[4];
  const float* bqkv   = (const float*)d_in[5];
  const float* Wproj  = (const float*)d_in[6];
  const float* bproj  = (const float*)d_in[7];
  const int ncu = in_sizes[1];
  (void)n_in; (void)out_size; (void)ws_size;

  char* ws = (char*)d_ws;
  size_t off = 0;
  auto carve = [&](size_t bytes) -> void* {
    void* p = ws + off;
    off = (off + bytes + 255) & ~(size_t)255;
    return p;
  };
  bf16*  hbf    = (bf16*)carve((size_t)kN * kDim * sizeof(bf16));
  bf16*  wqkvT  = (bf16*)carve((size_t)kNqkv * kDim * sizeof(bf16));
  bf16*  wprojT = (bf16*)carve((size_t)kDim * kDim * sizeof(bf16));
  float* qkv    = (float*)carve((size_t)kN * kNqkv * sizeof(float));
  bf16*  Qb     = (bf16*)carve((size_t)kH * kN * kDp * sizeof(bf16));
  bf16*  Kb     = (bf16*)carve((size_t)kH * kN * kDp * sizeof(bf16));
  bf16*  Vt     = (bf16*)carve((size_t)kH * kHd * kN * sizeof(bf16));
  bf16*  Ob     = (bf16*)carve((size_t)kN * kDim * sizeof(bf16));

  cvt_bf16_kernel<<<(kN * kDim + 255) / 256, 256, 0, stream>>>(hidden, hbf, kN * kDim);
  transpose_bf16_kernel<<<(kDim * kNqkv + 255) / 256, 256, 0, stream>>>(Wqkv, wqkvT,
                                                                        kDim, kNqkv);
  transpose_bf16_kernel<<<(kDim * kDim + 255) / 256, 256, 0, stream>>>(Wproj, wprojT,
                                                                       kDim, kDim);

  // QKV projection: 2048x1280 @ 1280x3840 (+bias) -> f32 qkv
  gemm_bf16_kernel<<<(kN / 32) * (kNqkv / 64) / 8, 256, 0, stream>>>(
      hbf, wqkvT, bqkv, qkv, kN, kNqkv, kDim);

  rope_pack_kernel<<<kN * kH, kDp, 0, stream>>>(qkv, cosb, sinb, Qb, Kb, Vt);

  attn_kernel<<<kH * (kN / 64), 128, 0, stream>>>(Qb, Kb, Vt, cu, ncu, Ob);

  // Output projection: 2048x1280 @ 1280x1280 (+bias) -> f32 d_out
  gemm_bf16_kernel<<<(kN / 32) * (kDim / 64) / 8, 256, 0, stream>>>(
      Ob, wprojT, bproj, (float*)d_out, kN, kDim, kDim);
}
